// NetTransform_721554505793
// MI455X (gfx1250) — matrix operand order repaired and checked by reading, not compiled, not run on hardware
//
#include <hip/hip_runtime.h>

// ---------------------------------------------------------------------------
// Problem constants (from reference): B=8, H=1, T=2048, C=512, fp32 in/out.
// Outputs (flat, concatenated): c[B,T,C], kernel[B,T,T], score[B,T,T],
// kernel_kk[B,T,T], kernel_target[B,T,T].
// ---------------------------------------------------------------------------
#define B_   8
#define T_   2048
#define C_   512
#define MINV (-3.4028234663852886e38f)
#define SCALE 0.04419417382415922f   // 1/sqrt(512)

typedef __attribute__((ext_vector_type(16))) __bf16 v16bf;
typedef __attribute__((ext_vector_type(8)))  float  v8f;
typedef __attribute__((address_space(3))) unsigned short lds_us;

union Frag { v16bf v; unsigned u[8]; };

// f32 -> bf16 round-to-nearest-even
__device__ __forceinline__ unsigned short f2bf(float f) {
  unsigned u = __float_as_uint(f);
  u += 0x7FFFu + ((u >> 16) & 1u);
  return (unsigned short)(u >> 16);
}
__device__ __forceinline__ unsigned pack2(float a, float b) {
  return (unsigned)f2bf(a) | ((unsigned)f2bf(b) << 16);
}

#define KBLK 32
#define LDT  40   // LDS row stride in ushort (32 data + 8 pad; 80B = 5*16B)

// ---------------------------------------------------------------------------
// Kernel 0: one-shot f32 -> bf16 conversion of q and k into workspace.
// ---------------------------------------------------------------------------
__global__ __launch_bounds__(256) void cvt_bf16_kernel(
    const float* __restrict__ q, const float* __restrict__ k,
    unsigned short* __restrict__ qbf, unsigned short* __restrict__ kbf)
{
  const size_t idx = ((size_t)blockIdx.x * 256 + threadIdx.x) * 8;
  const float* src = (blockIdx.y == 0) ? q : k;
  unsigned short* dst = (blockIdx.y == 0) ? qbf : kbf;
  const float4* s4 = (const float4*)(src + idx);
  float4 a = s4[0], b = s4[1];
  uint4 o;
  o.x = pack2(a.x, a.y); o.y = pack2(a.z, a.w);
  o.z = pack2(b.x, b.y); o.w = pack2(b.z, b.w);
  *(uint4*)(dst + idx) = o;
}

// ---------------------------------------------------------------------------
// Kernel 1: score = mask(k.q^T / sqrt(C)),  kernel_kk = k.k^T  (shared A = k)
// Block = 256 threads (8 waves), 128x128 tile of both outputs.
// Staging uses CDNA5 async global->LDS b128 copies, double-buffered (ASYNCcnt).
// ---------------------------------------------------------------------------
__global__ __launch_bounds__(256) void score_kk_kernel(
    const unsigned short* __restrict__ qbf, const unsigned short* __restrict__ kbf,
    const int* __restrict__ ilens,
    float* __restrict__ score, float* __restrict__ kk)
{
  // [2 buffers][3 tiles: A=k(m), Bq=q(n), Bk=k(n)][128 rows * LDT]
  __shared__ __align__(16) unsigned short lds[2][3][128 * LDT];

  const int bz  = blockIdx.z;
  const int m0  = blockIdx.y * 128;
  const int n0  = blockIdx.x * 128;
  const int tid = threadIdx.x;
  const int ilen = ilens[bz];

  const unsigned short* kbA = kbf + (size_t)bz * T_ * C_ + (size_t)m0 * C_;
  const unsigned short* qbB = qbf + (size_t)bz * T_ * C_ + (size_t)n0 * C_;
  const unsigned short* kbB = kbf + (size_t)bz * T_ * C_ + (size_t)n0 * C_;

  // Per-thread async-copy assignment: row = tid>>1, 32-byte chunk per thread,
  // issued as 2x b128. Global offset (bytes) relative to (row0, c0) corner.
  const int arow = tid >> 1;
  const unsigned ch = (unsigned)(tid & 1) * 32;
  const unsigned voff = (unsigned)arow * (C_ * 2) + ch;

  const int wave = tid >> 5, lane = tid & 31;
  const int wr = wave >> 1;           // 0..3 : 32-row strip
  const int wc = wave & 1;            // 0..1 : 64-col strip
  const int nlan = lane & 15, hlan = lane >> 4;

#define ISSUE_TILE(bufi, tile, gb)                                            \
  {                                                                           \
    lds_us* lp = (lds_us*)(&lds[bufi][tile][arow * LDT] ) + (ch >> 1);        \
    asm volatile("global_load_async_to_lds_b128 %0, %1, %2"                   \
                 :: "v"(lp), "v"(voff), "s"(gb) : "memory");                  \
    asm volatile("global_load_async_to_lds_b128 %0, %1, %2 offset:16"         \
                 :: "v"(lp), "v"(voff), "s"(gb) : "memory");                  \
  }

  v8f accS[2][4] = {};
  v8f accK[2][4] = {};

  // Prologue: fill buffer 0 for c0 = 0.
  ISSUE_TILE(0, 0, kbA); ISSUE_TILE(0, 1, qbB); ISSUE_TILE(0, 2, kbB);

  for (int step = 0; step < C_ / KBLK; ++step) {
    const int p = step & 1;
    // My async loads into buf p are done; then barrier => everyone's are, and
    // everyone finished reading buf p^1 last iteration.
    asm volatile("s_wait_asynccnt 0x0" ::: "memory");
    __syncthreads();
    if (step + 1 < C_ / KBLK) {
      const int c0n = (step + 1) * KBLK;     // ushort offset == folded via saddr
      ISSUE_TILE(p ^ 1, 0, kbA + c0n);
      ISSUE_TILE(p ^ 1, 1, qbB + c0n);
      ISSUE_TILE(p ^ 1, 2, kbB + c0n);
    }

    // ---- A fragments: k rows (16x32, contiguous k = 16h + s) ----
    Frag A[2];
#pragma unroll
    for (int r = 0; r < 2; ++r) {
      const unsigned* rowA = (const unsigned*)&lds[p][0][((wr * 2 + r) * 16 + nlan) * LDT];
#pragma unroll
      for (int v = 0; v < 4; ++v) {
        A[r].u[v]     = rowA[hlan * 8 + v];        // K = h*16 + 2v (+j)
        A[r].u[4 + v] = rowA[hlan * 8 + 4 + v];    // K = h*16 + 8 + 2v
      }
    }
    // ---- B fragments + WMMA ----
#pragma unroll
    for (int c = 0; c < 4; ++c) {
      const int nsub = wc * 4 + c;
      const unsigned* rowQ = (const unsigned*)&lds[p][1][(nsub * 16 + nlan) * LDT] + hlan * 8;
      const unsigned* rowK = (const unsigned*)&lds[p][2][(nsub * 16 + nlan) * LDT] + hlan * 8;
      Frag Bq, Bk;
#pragma unroll
      for (int v = 0; v < 8; ++v) { Bq.u[v] = rowQ[v]; Bk.u[v] = rowK[v]; }
#pragma unroll
      for (int r = 0; r < 2; ++r) {
        accS[r][c] = __builtin_amdgcn_wmma_f32_16x16x32_bf16(
            false, A[r].v, false, Bq.v, (short)0, accS[r][c], false, false);
        accK[r][c] = __builtin_amdgcn_wmma_f32_16x16x32_bf16(
            false, A[r].v, false, Bk.v, (short)0, accK[r][c], false, false);
      }
    }
  }
#undef ISSUE_TILE

  // ---- epilogue: C/D layout lane<16 -> M=e, lane>=16 -> M=e+8; N=lane&15 ----
  float* sb  = score + (size_t)bz * T_ * T_;
  float* kkb = kk    + (size_t)bz * T_ * T_;
#pragma unroll
  for (int r = 0; r < 2; ++r) {
#pragma unroll
    for (int c = 0; c < 4; ++c) {
      const int gn = n0 + (wc * 4 + c) * 16 + nlan;
#pragma unroll
      for (int e = 0; e < 8; ++e) {
        const int gm = m0 + (wr * 2 + r) * 16 + hlan * 8 + e;
        float sv = accS[r][c][e] * SCALE;
        if (gm == gn || gm >= ilen || gn >= ilen) sv = MINV;
        sb [(size_t)gm * T_ + gn] = sv;
        kkb[(size_t)gm * T_ + gn] = accK[r][c][e];
      }
    }
  }
}

// ---------------------------------------------------------------------------
// Kernel 2: row softmax with masked zeroing. One block per (b,t) row.
// ---------------------------------------------------------------------------
__global__ __launch_bounds__(256) void softmax_kernel(
    const float* __restrict__ score, const int* __restrict__ ilens,
    float* __restrict__ kern)
{
  __shared__ float red[256];
  const int t = blockIdx.x, b = blockIdx.y, tid = threadIdx.x;
  const int ilen = ilens[b];
  const float* row = score + ((size_t)b * T_ + t) * T_;

  float x[8], mx = MINV;
#pragma unroll
  for (int i = 0; i < 8; ++i) { x[i] = row[tid + i * 256]; mx = fmaxf(mx, x[i]); }
  red[tid] = mx; __syncthreads();
  for (int s = 128; s > 0; s >>= 1) {
    if (tid < s) red[tid] = fmaxf(red[tid], red[tid + s]);
    __syncthreads();
  }
  mx = red[0]; __syncthreads();

  float ex[8], sum = 0.f;
#pragma unroll
  for (int i = 0; i < 8; ++i) { ex[i] = __expf(x[i] - mx); sum += ex[i]; }
  red[tid] = sum; __syncthreads();
  for (int s = 128; s > 0; s >>= 1) {
    if (tid < s) red[tid] += red[tid + s];
    __syncthreads();
  }
  const float inv = 1.f / red[0];

  float* orow = kern + ((size_t)b * T_ + t) * T_;
#pragma unroll
  for (int i = 0; i < 8; ++i) {
    const int s = tid + i * 256;
    float v = ex[i] * inv;
    if (s == t || s >= ilen || t >= ilen) v = 0.f;
    orow[s] = v;
  }
}

// ---------------------------------------------------------------------------
// Kernel 3: c = kernel . q   (A = kernel[t,s] f32->bf16, B = q[s,c])
// q staged transposed in LDS so B-fragment reads are contiguous per lane.
// ---------------------------------------------------------------------------
__global__ __launch_bounds__(256) void ctx_kernel(
    const float* __restrict__ kern, const float* __restrict__ q,
    float* __restrict__ cout)
{
  __shared__ __align__(16) unsigned short lA [128 * LDT];  // kernel rows
  __shared__ __align__(16) unsigned short lBt[128 * LDT];  // q^T: [channel][s]

  const int bz  = blockIdx.z;
  const int m0  = blockIdx.y * 128;   // t rows
  const int n0  = blockIdx.x * 128;   // channel cols
  const int tid = threadIdx.x;

  const float* kb = kern + (size_t)bz * T_ * T_;
  const float* qb = q    + (size_t)bz * T_ * C_;

  const int lrow = tid >> 1, lcol = (tid & 1) * 16;
  const int srow = tid >> 3, scb  = (tid & 7) * 16;   // B-tile: 32 s-rows x 128 cols

  const int wave = tid >> 5, lane = tid & 31;
  const int wr = wave >> 1, wc = wave & 1;
  const int nlan = lane & 15, hlan = lane >> 4;

  v8f acc[2][4] = {};

  for (int s0 = 0; s0 < T_; s0 += KBLK) {
    __syncthreads();
    {
      // A: kernel[m0+lrow, s0+lcol .. +15]
      const float4* sA = (const float4*)(kb + (size_t)(m0 + lrow) * T_ + s0 + lcol);
      unsigned* dA = (unsigned*)&lA[lrow * LDT + lcol];
#pragma unroll
      for (int i = 0; i < 4; ++i) {
        float4 a = sA[i]; dA[2*i] = pack2(a.x, a.y); dA[2*i+1] = pack2(a.z, a.w);
      }
      // B: q[s0+srow, n0+scb .. +15] stored transposed: lBt[channel][s]
      const float4* sB = (const float4*)(qb + (size_t)(s0 + srow) * C_ + n0 + scb);
#pragma unroll
      for (int i = 0; i < 4; ++i) {
        float4 b = sB[i];
        lBt[(scb + 4*i + 0) * LDT + srow] = f2bf(b.x);
        lBt[(scb + 4*i + 1) * LDT + srow] = f2bf(b.y);
        lBt[(scb + 4*i + 2) * LDT + srow] = f2bf(b.z);
        lBt[(scb + 4*i + 3) * LDT + srow] = f2bf(b.w);
      }
    }
    __syncthreads();

    Frag A[2];
#pragma unroll
    for (int r = 0; r < 2; ++r) {
      const unsigned* rowA = (const unsigned*)&lA[((wr * 2 + r) * 16 + nlan) * LDT];
#pragma unroll
      for (int v = 0; v < 4; ++v) {
        A[r].u[v]     = rowA[hlan * 8 + v];
        A[r].u[4 + v] = rowA[hlan * 8 + 4 + v];
      }
    }
#pragma unroll
    for (int c = 0; c < 4; ++c) {
      const int nsub = wc * 4 + c;
      const unsigned* rowB = (const unsigned*)&lBt[(nsub * 16 + nlan) * LDT] + hlan * 8;
      Frag Bf;
#pragma unroll
      for (int v = 0; v < 8; ++v) Bf.u[v] = rowB[v];
#pragma unroll
      for (int r = 0; r < 2; ++r)
        acc[r][c] = __builtin_amdgcn_wmma_f32_16x16x32_bf16(
            false, A[r].v, false, Bf.v, (short)0, acc[r][c], false, false);
    }
  }

  float* cb = cout + (size_t)bz * T_ * C_;
#pragma unroll
  for (int r = 0; r < 2; ++r) {
#pragma unroll
    for (int c = 0; c < 4; ++c) {
      const int gn = n0 + (wc * 4 + c) * 16 + nlan;
#pragma unroll
      for (int e = 0; e < 8; ++e) {
        const int gm = m0 + (wr * 2 + r) * 16 + hlan * 8 + e;
        cb[(size_t)gm * C_ + gn] = acc[r][c][e];
      }
    }
  }
}

// ---------------------------------------------------------------------------
// Kernel 4a: x[t] = (t==0) ? 1 : 1 - kernel[b,t-1,t]; 64-element chunk products
// ---------------------------------------------------------------------------
__global__ __launch_bounds__(256) void energy_kernel(
    const float* __restrict__ kern, float* __restrict__ xbuf,
    float* __restrict__ cp)
{
  __shared__ float xs[T_];
  const int b = blockIdx.x, tid = threadIdx.x;
  const float* kb = kern + (size_t)b * T_ * T_;
  for (int t = tid; t < T_; t += 256) {
    float v = (t == 0) ? 1.f : 1.f - kb[(size_t)(t - 1) * T_ + t];
    xs[t] = v;
    xbuf[b * T_ + t] = v;
  }
  __syncthreads();
  if (tid < 32) {
    float p = 1.f;
#pragma unroll 8
    for (int i = 0; i < 64; ++i) p *= xs[tid * 64 + i];
    cp[b * 32 + tid] = p;
  }
}

// ---------------------------------------------------------------------------
// Kernel 4b: kernel_target[i,j] = 0 (j>i), 1 (j==i), prod_{r=j..i-1} x[r] (j<i)
// 64x64 tiles; off-diagonal tiles factor as cp[j] * M * rp[i].
// ---------------------------------------------------------------------------
__global__ __launch_bounds__(256) void fb_kernel(
    const float* __restrict__ xbuf, const float* __restrict__ cp,
    float* __restrict__ kt)
{
  const int tj = blockIdx.x, ti = blockIdx.y, b = blockIdx.z;
  const int tid = threadIdx.x;
  float* out = kt + (size_t)b * T_ * T_;
  const int i0 = ti * 64, j0 = tj * 64;

  if (tj > ti) {                       // strictly above diagonal: zeros
#pragma unroll
    for (int kq = 0; kq < 16; ++kq) {
      int p = kq * 256 + tid, i = p >> 6, j = p & 63;
      out[(size_t)(i0 + i) * T_ + j0 + j] = 0.f;
    }
    return;
  }

  __shared__ float xi[64], xj[64], rp[64], cpx[64];
  const float* xb = xbuf + b * T_;
  if (tid < 64) { xi[tid] = xb[i0 + tid]; xj[tid] = xb[j0 + tid]; }
  __syncthreads();

  if (tj == ti) {                      // diagonal tile: in-tile products
    for (int kq = 0; kq < 16; ++kq) {
      int p = kq * 256 + tid, i = p >> 6, j = p & 63;
      float v;
      if (j > i)       v = 0.f;
      else if (j == i) v = 1.f;
      else { v = 1.f; for (int r = j; r < i; ++r) v *= xi[r]; }
      out[(size_t)(i0 + i) * T_ + j0 + j] = v;
    }
  } else {                             // below diagonal: cp[j] * M * rp[i]
    if (tid < 64) {
      float p = 1.f;                   // prod x[i0 .. i0+tid-1]
      for (int r = 0; r < tid; ++r) p *= xi[r];
      rp[tid] = p;
      float s = 1.f;                   // prod x[j0+tid .. j0+63]
      for (int r = tid; r < 64; ++r) s *= xj[r];
      cpx[tid] = s;
    }
    __syncthreads();
    float M = 1.f;
    for (int c = tj + 1; c < ti; ++c) M *= cp[b * 32 + c];
#pragma unroll
    for (int kq = 0; kq < 16; ++kq) {
      int p = kq * 256 + tid, i = p >> 6, j = p & 63;
      out[(size_t)(i0 + i) * T_ + j0 + j] = cpx[j] * M * rp[i];
    }
  }
}

// ---------------------------------------------------------------------------
extern "C" void kernel_launch(void* const* d_in, const int* in_sizes, int n_in,
                              void* d_out, int out_size, void* d_ws, size_t ws_size,
                              hipStream_t stream) {
  const float* q     = (const float*)d_in[0];
  const float* k     = (const float*)d_in[1];
  const int*   ilens = (const int*)d_in[2];
  float* out = (float*)d_out;

  const size_t NC = (size_t)B_ * T_ * C_;   // 8,388,608
  const size_t NT = (size_t)B_ * T_ * T_;   // 33,554,432
  float* c_out = out;
  float* kern  = out + NC;
  float* score = out + NC + NT;
  float* kk    = out + NC + 2 * NT;
  float* kt    = out + NC + 3 * NT;

  // Workspace: q_bf16 | k_bf16 | xbuf | chunk-products
  unsigned short* qbf = (unsigned short*)d_ws;
  unsigned short* kbf = qbf + NC;
  float* xbuf = (float*)(kbf + NC);         // B*T floats
  float* cp   = xbuf + B_ * T_;             // B*32 floats

  dim3 blk(256);
  cvt_bf16_kernel<<<dim3((unsigned)(NC / (256 * 8)), 2), blk, 0, stream>>>(q, k, qbf, kbf);
  score_kk_kernel<<<dim3(T_ / 128, T_ / 128, B_), blk, 0, stream>>>(qbf, kbf, ilens, score, kk);
  softmax_kernel <<<dim3(T_, B_),                 blk, 0, stream>>>(score, ilens, kern);
  ctx_kernel     <<<dim3(C_ / 128, T_ / 128, B_), blk, 0, stream>>>(kern, q, c_out);
  energy_kernel  <<<dim3(B_),                     blk, 0, stream>>>(kern, xbuf, cp);
  fb_kernel      <<<dim3(T_ / 64, T_ / 64, B_),   blk, 0, stream>>>(xbuf, cp, kt);
}
